// MultiScaleCompressedAttention_73332271612355
// MI455X (gfx1250) — matrix operand order repaired and checked by reading, not compiled, not run on hardware
//
#include <hip/hip_runtime.h>
#include <math.h>

// ---------------- problem constants ----------------
#define BATCH 2
#define LSEQ  2048
#define EMB   1024
#define NH    16
#define HD    64
#define NTOK  (BATCH * LSEQ)      // 4096
#define BH    (BATCH * NH)        // 32
#define BHLD  ((size_t)BH * LSEQ * HD)  // 4194304

typedef __attribute__((ext_vector_type(16))) _Float16 v16h;
typedef __attribute__((ext_vector_type(8)))  float    v8f;

union U8H { uint4 v; _Float16 h[8]; };

#define NEG_INF (-__builtin_inff())

// ---------------- WMMA fragment helpers (gfx1250 wave32 layouts) ----------------

// A: 16x32 f16, row-major source with pitch lda.
// lane l: row m = l&15, half = l>>4; a[i]=A[m][k0+half*8+i], a[8+i]=A[m][k0+16+half*8+i]
__device__ inline v16h load_a16(const _Float16* A, int lda, int m0, int k0) {
  int lane = threadIdx.x & 31;
  int m = lane & 15, hf = (lane >> 4) & 1;
  const _Float16* p = A + (size_t)(m0 + m) * lda + k0 + hf * 8;
  U8H lo, hi;
  lo.v = *(const uint4*)p;
  hi.v = *(const uint4*)(p + 16);
  v16h a;
#pragma unroll
  for (int i = 0; i < 8; ++i) { a[i] = lo.h[i]; a[8 + i] = hi.h[i]; }
  return a;
}

// B: 32x16 f16 for C = A*B with B[k][n] = Brow[n][k] (NT gemm).
// lane l: col n = l&15, half = l>>4; b[i]=Brow[n0+n][k0+half*8+i], b[8+i]=Brow[n0+n][k0+16+half*8+i]
__device__ inline v16h load_b16(const _Float16* Brow, int ldb, int n0, int k0) {
  int lane = threadIdx.x & 31;
  int n = lane & 15, hf = (lane >> 4) & 1;
  const _Float16* p = Brow + (size_t)(n0 + n) * ldb + k0 + hf * 8;
  U8H lo, hi;
  lo.v = *(const uint4*)p;
  hi.v = *(const uint4*)(p + 16);
  v16h b;
#pragma unroll
  for (int i = 0; i < 8; ++i) { b[i] = lo.h[i]; b[8 + i] = hi.h[i]; }
  return b;
}

__device__ inline v8f wmma16(v16h a, v16h b, v8f c) {
  return __builtin_amdgcn_wmma_f32_16x16x32_f16(false, a, false, b, (short)0, c,
                                                false, false);
}

__device__ inline void store_c(float* C, int ldc, int m0, int n0, v8f c) {
  int lane = threadIdx.x & 31;
  int n = lane & 15, hf = (lane >> 4) & 1;
#pragma unroll
  for (int r = 0; r < 8; ++r)
    C[(size_t)(m0 + r + 8 * hf) * ldc + n0 + n] = c[r];
}

__device__ inline float rowred_max16(float v) {
#pragma unroll
  for (int m = 1; m < 16; m <<= 1) v = fmaxf(v, __shfl_xor(v, m, 32));
  return v;
}
__device__ inline float rowred_sum16(float v) {
#pragma unroll
  for (int m = 1; m < 16; m <<= 1) v += __shfl_xor(v, m, 32);
  return v;
}

// ---------------- elementwise / small kernels ----------------

__global__ void f32_to_f16_kernel(const float* in, _Float16* out, int n) {
  int i = blockIdx.x * blockDim.x + threadIdx.x;
  if (i < n) out[i] = (_Float16)in[i];
}

__global__ void __launch_bounds__(256)
ln_kernel(const float* __restrict__ in, const float* __restrict__ g,
          const float* __restrict__ b, float* __restrict__ o32,
          _Float16* __restrict__ o16) {
  __shared__ float red[256];
  int tok = blockIdx.x, tid = threadIdx.x;
  const float* x = in + (size_t)tok * EMB;
  float s = 0.f;
  for (int k = tid; k < EMB; k += 256) s += x[k];
  red[tid] = s; __syncthreads();
  for (int st = 128; st > 0; st >>= 1) { if (tid < st) red[tid] += red[tid + st]; __syncthreads(); }
  float mu = red[0] * (1.f / EMB);
  __syncthreads();
  float v = 0.f;
  for (int k = tid; k < EMB; k += 256) { float d = x[k] - mu; v += d * d; }
  red[tid] = v; __syncthreads();
  for (int st = 128; st > 0; st >>= 1) { if (tid < st) red[tid] += red[tid + st]; __syncthreads(); }
  float inv = rsqrtf(red[0] * (1.f / EMB) + 1e-5f);
  for (int k = tid; k < EMB; k += 256) {
    float y = (x[k] - mu) * inv * g[k] + b[k];
    if (o32) o32[(size_t)tok * EMB + k] = y;
    o16[(size_t)tok * EMB + k] = (_Float16)y;
  }
}

// qkv [NTOK,3072] -> head-major f32 q/k/v, f16 q/k, f16 V^T
__global__ void qkv_transform_kernel(const float* __restrict__ qkv,
                                     float* qf, float* kf, float* vf,
                                     _Float16* qh, _Float16* kh, _Float16* vth) {
  int i = blockIdx.x * blockDim.x + threadIdx.x;  // over NTOK*EMB
  int tok = i >> 10, e = i & 1023;
  int h = e >> 6, d = e & 63;
  int bb = tok >> 11, l = tok & (LSEQ - 1);
  float qv = qkv[(size_t)tok * 3072 + e];
  float kv = qkv[(size_t)tok * 3072 + 1024 + e];
  float vv = qkv[(size_t)tok * 3072 + 2048 + e];
  size_t bh = (size_t)(bb * NH + h);
  size_t hm = (bh * LSEQ + l) * HD + d;
  qf[hm] = qv; kf[hm] = kv; vf[hm] = vv;
  qh[hm] = (_Float16)qv; kh[hm] = (_Float16)kv;
  vth[(bh * HD + d) * LSEQ + l] = (_Float16)vv;
}

// ---------------- generic WMMA NT GEMM: C[M,N] = A[M,K] * B[N,K]^T ----------------
// block = 128 threads (4 waves); wave computes one 16x64 tile.
__global__ void __launch_bounds__(128)
gemm_nt_f16_kernel(const _Float16* __restrict__ A, const _Float16* __restrict__ B,
                   float* __restrict__ C, int M, int N, int K) {
  int w = threadIdx.x >> 5;
  int m0 = blockIdx.x * 16;
  int n0 = (blockIdx.y * 4 + w) * 64;
  v8f c[4];
#pragma unroll
  for (int j = 0; j < 4; ++j)
#pragma unroll
    for (int r = 0; r < 8; ++r) c[j][r] = 0.f;
  for (int k0 = 0; k0 < K; k0 += 32) {
    v16h a = load_a16(A, K, m0, k0);
#pragma unroll
    for (int j = 0; j < 4; ++j) {
      v16h b = load_b16(B, K, n0 + 16 * j, k0);
      c[j] = wmma16(a, b, c[j]);
    }
  }
#pragma unroll
  for (int j = 0; j < 4; ++j) store_c(C, N, m0, n0 + 16 * j, c[j]);
}

// ---------------- flash-style causal attention (WMMA) ----------------
// one wave per (b,h, 16-query tile); keys streamed 32 at a time.
__global__ void __launch_bounds__(128)
flash_attn_kernel(const _Float16* __restrict__ qh, const _Float16* __restrict__ kh,
                  const _Float16* __restrict__ vth, float* __restrict__ glb) {
  __shared__ __align__(16) _Float16 Plds[4][16][32];
  int w = threadIdx.x >> 5, lane = threadIdx.x & 31;
  int wg = blockIdx.x * 4 + w;
  int bh = wg >> 7, qt = wg & 127;
  int q0 = qt * 16;
  const _Float16* Q  = qh  + (size_t)bh * LSEQ * HD;
  const _Float16* Kp = kh  + (size_t)bh * LSEQ * HD;
  const _Float16* Vt = vth + (size_t)bh * HD * LSEQ;

  v16h aq0 = load_a16(Q, HD, q0, 0);
  v16h aq1 = load_a16(Q, HD, q0, 32);

  float rm[8], rs[8];
  v8f oc[4];
#pragma unroll
  for (int r = 0; r < 8; ++r) { rm[r] = NEG_INF; rs[r] = 0.f; }
#pragma unroll
  for (int j = 0; j < 4; ++j)
#pragma unroll
    for (int r = 0; r < 8; ++r) oc[j][r] = 0.f;

  v8f zf;
#pragma unroll
  for (int r = 0; r < 8; ++r) zf[r] = 0.f;

  int n = lane & 15, hf = (lane >> 4) & 1;
  const float scale = 0.125f;  // 1/sqrt(64)
  int nkt = (q0 >> 5) + 1;

  for (int kt = 0; kt < nkt; ++kt) {
    int kt0 = kt * 32;
    v8f s0 = wmma16(aq0, load_b16(Kp, HD, kt0, 0), zf);
    s0 = wmma16(aq1, load_b16(Kp, HD, kt0, 32), s0);
    v8f s1 = wmma16(aq0, load_b16(Kp, HD, kt0 + 16, 0), zf);
    s1 = wmma16(aq1, load_b16(Kp, HD, kt0 + 16, 32), s1);

    float alpha[8], p0[8], p1[8];
#pragma unroll
    for (int r = 0; r < 8; ++r) {
      int q = q0 + r + 8 * hf;
      float x0 = (kt0 + n      <= q) ? s0[r] * scale : NEG_INF;
      float x1 = (kt0 + 16 + n <= q) ? s1[r] * scale : NEG_INF;
      float t = rowred_max16(fmaxf(x0, x1));
      float mn = fmaxf(rm[r], t);
      alpha[r] = __expf(rm[r] - mn);
      p0[r] = __expf(x0 - mn);
      p1[r] = __expf(x1 - mn);
      float ls = rowred_sum16(p0[r] + p1[r]);
      rs[r] = rs[r] * alpha[r] + ls;
      rm[r] = mn;
    }
#pragma unroll
    for (int j = 0; j < 4; ++j)
#pragma unroll
      for (int r = 0; r < 8; ++r) oc[j][r] *= alpha[r];

    // redistribute P (C layout) -> A layout via per-wave LDS
#pragma unroll
    for (int r = 0; r < 8; ++r) {
      Plds[w][r + 8 * hf][n]      = (_Float16)p0[r];
      Plds[w][r + 8 * hf][16 + n] = (_Float16)p1[r];
    }
    asm volatile("s_wait_dscnt 0" ::: "memory");
    v16h ap = load_a16(&Plds[w][0][0], 32, 0, 0);
    asm volatile("" ::: "memory");
#pragma unroll
    for (int j = 0; j < 4; ++j)
      oc[j] = wmma16(ap, load_b16(Vt, LSEQ, j * 16, kt0), oc[j]);
  }
#pragma unroll
  for (int j = 0; j < 4; ++j)
#pragma unroll
    for (int r = 0; r < 8; ++r)
      glb[(size_t)bh * LSEQ * HD + (size_t)(q0 + r + 8 * hf) * HD + j * 16 + n] =
          oc[j][r] / rs[r];
}

// ---------------- compression MLP: out = W2 * gelu(W1 * x + b1) + b2 ----------------
// grid = BH*Kc blocks, block = 32 threads.
__global__ void __launch_bounds__(32)
comp_mlp_kernel(const float* __restrict__ src, const float* __restrict__ w1,
                const float* __restrict__ b1, const float* __restrict__ w2,
                const float* __restrict__ b2, float* __restrict__ dst,
                int blk, int Kc) {
  int g = blockIdx.x;
  int kc = g % Kc, bh = g / Kc;
  int t = threadIdx.x;
  __shared__ float hid[32];
  const float* base = src + (size_t)bh * LSEQ * HD + (size_t)kc * blk * HD;
  int inlen = blk * 32;
  const float* w1r = w1 + (size_t)t * inlen;
  float acc = b1[t];
  for (int j = 0; j < blk; ++j) {
    const float* xr = base + (size_t)j * HD;  // first 32 dims (lora half)
    const float* wr = w1r + j * 32;
#pragma unroll
    for (int d = 0; d < 32; ++d) acc += wr[d] * xr[d];
  }
  // quick_gelu(x) = x * sigmoid(1.702x)
  hid[t] = acc / (1.f + __expf(-1.702f * acc));
  __syncthreads();
  float o = b2[t];
  const float* w2r = w2 + t * 32;
#pragma unroll
  for (int i = 0; i < 32; ++i) o += w2r[i] * hid[i];
  dst[((size_t)bh * Kc + kc) * 32 + t] = o;
}

__device__ inline float dot32s(const float* a, const float* b) {
  float s = 0.f;
#pragma unroll
  for (int i = 0; i < 32; ++i) s += a[i] * b[i];
  return s;
}

// ---------------- compressed attention + block scores ----------------
// grid (LSEQ/256, BH), block 256; one thread per query; dyn-shared: Kc*32 + Kc floats.
__global__ void __launch_bounds__(256)
comp_attn_kernel(const float* __restrict__ qf, const float* __restrict__ kcomp,
                 const float* __restrict__ vcomp, float* __restrict__ outbuf,
                 float* __restrict__ bscore, int blk, int Kc) {
  extern __shared__ float smem[];
  float* kl  = smem;            // Kc*32
  float* bsl = smem + Kc * 32;  // Kc
  int bh = blockIdx.y, tid = threadIdx.x;
  for (int i = tid; i < Kc * 32; i += 256) kl[i] = kcomp[(size_t)bh * Kc * 32 + i];
  for (int i = tid; i < Kc; i += 256) bsl[i] = 0.f;
  __syncthreads();

  int q = blockIdx.x * 256 + tid;
  float qv[32];
  const float* qp = qf + (size_t)bh * LSEQ * HD + (size_t)q * HD;  // q1 = dims [0,32)
#pragma unroll
  for (int d = 0; d < 32; ++d) qv[d] = qp[d];
  int qb = q / blk;
  const float scale = 0.1767766952966369f;  // 1/sqrt(32)

  float m = NEG_INF, s = 0.f;
  for (int kc = 0; kc <= qb; ++kc) {
    float v = dot32s(qv, &kl[kc * 32]) * scale;
    float mn = fmaxf(m, v);
    s = s * __expf(m - mn) + __expf(v - mn);
    m = mn;
  }
  float inv = 1.f / s;
  float out[32];
#pragma unroll
  for (int d = 0; d < 32; ++d) out[d] = 0.f;
  const float* vcb = vcomp + (size_t)bh * Kc * 32;
  for (int kc = 0; kc <= qb; ++kc) {
    float v = dot32s(qv, &kl[kc * 32]) * scale;
    float e = __expf(v - m) * inv;
    const float* vr = vcb + kc * 32;
#pragma unroll
    for (int d = 0; d < 32; ++d) out[d] += e * vr[d];
    atomicAdd(&bsl[kc], e);
  }
  float* op = outbuf + (size_t)bh * LSEQ * HD + (size_t)q * HD;  // comp_out -> dims [0,32)
#pragma unroll
  for (int d = 0; d < 32; ++d) op[d] = out[d];
  __syncthreads();
  for (int i = tid; i < Kc; i += 256) atomicAdd(&bscore[bh * Kc + i], bsl[i]);
}

// ---------------- top-k (matches jax.lax.top_k tie-breaking) ----------------
__global__ void topk_kernel(const float* __restrict__ bscore, int* __restrict__ idx,
                            int Kc, int K) {
  if (threadIdx.x != 0) return;
  int bh = blockIdx.x;
  const float* bs = bscore + bh * Kc;
  bool used[256];
  for (int i = 0; i < Kc; ++i) used[i] = false;
  for (int t = 0; t < K; ++t) {
    int best = 0; float bv = NEG_INF;
    for (int i = 0; i < Kc; ++i)
      if (!used[i] && bs[i] > bv) { bv = bs[i]; best = i; }
    used[best] = true;
    idx[bh * K + t] = best;
  }
}

// ---------------- selected-block attention (top_k*blk == 64) ----------------
// grid (LSEQ/256, BH), block 256.
__global__ void __launch_bounds__(256)
sel_attn_kernel(const float* __restrict__ qf, const float* __restrict__ kf,
                const float* __restrict__ vf, const int* __restrict__ idx,
                float* __restrict__ outbuf, int blk, int K) {
  __shared__ float ksel[64 * 32];
  __shared__ float vsel[64 * 32];
  __shared__ int bidx[8];
  int bh = blockIdx.y, tid = threadIdx.x;
  if (tid < K) bidx[tid] = idx[bh * K + tid];
  __syncthreads();
  for (int i = tid; i < 64 * 32; i += 256) {
    int r = i >> 5, c = i & 31;
    int pos = bidx[r / blk] * blk + (r % blk);
    size_t off = (size_t)bh * LSEQ * HD + (size_t)pos * HD + 32 + c;  // k2/v2 half
    ksel[i] = kf[off];
    vsel[i] = vf[off];
  }
  __syncthreads();

  int q = blockIdx.x * 256 + tid;
  float qv[32];
  const float* qp = qf + (size_t)bh * LSEQ * HD + (size_t)q * HD + 32;  // q2
#pragma unroll
  for (int d = 0; d < 32; ++d) qv[d] = qp[d];
  int qb = q / blk;
  const float scale = 0.1767766952966369f;

  float m = NEG_INF, s = 0.f;
  for (int r = 0; r < 64; ++r) {
    if (qb >= bidx[r / blk]) {
      float v = dot32s(qv, &ksel[r * 32]) * scale;
      float mn = fmaxf(m, v);
      s = s * __expf(m - mn) + __expf(v - mn);
      m = mn;
    }
  }
  float inv = 1.f / s;
  float out[32];
#pragma unroll
  for (int d = 0; d < 32; ++d) out[d] = 0.f;
  for (int r = 0; r < 64; ++r) {
    float v = (qb >= bidx[r / blk]) ? dot32s(qv, &ksel[r * 32]) * scale : NEG_INF;
    float e = __expf(v - m) * inv;
    const float* vr = &vsel[r * 32];
#pragma unroll
    for (int d = 0; d < 32; ++d) out[d] += e * vr[d];
  }
  float* op = outbuf + (size_t)bh * LSEQ * HD + (size_t)q * HD + 32;  // slc_out dims
#pragma unroll
  for (int d = 0; d < 32; ++d) op[d] = out[d];
}

// ---------------- gates: sigmoid(x @ gate_w^T + gate_b) ----------------
__global__ void __launch_bounds__(64)
gate_kernel(const float* __restrict__ xln, const float* __restrict__ gw,
            const float* __restrict__ gb, float* __restrict__ gates) {
  int tok = blockIdx.x, t = threadIdx.x;
  if (t >= 48) return;
  const float* x = xln + (size_t)tok * EMB;
  const float* w = gw + (size_t)t * EMB;
  float a = gb[t];
  for (int k = 0; k < EMB; ++k) a += x[k] * w[k];
  gates[(size_t)tok * 48 + t] = 1.f / (1.f + __expf(-a));
}

// ---------------- gated combine -> (B,L,E) ----------------
__global__ void combine_kernel(const float* __restrict__ o1, const float* __restrict__ o2,
                               const float* __restrict__ og, const float* __restrict__ gates,
                               float* __restrict__ comb) {
  int i = blockIdx.x * blockDim.x + threadIdx.x;  // NTOK*EMB
  int tok = i >> 10, e = i & 1023;
  int h = e >> 6, d = e & 63;
  int bb = tok >> 11, l = tok & (LSEQ - 1);
  size_t hm = (((size_t)(bb * NH + h)) * LSEQ + l) * HD + d;
  const float* g = gates + (size_t)tok * 48 + h * 3;
  comb[i] = g[0] * o1[hm] + g[1] * o2[hm] + g[2] * og[hm];
}

// ---------------- host launch ----------------
extern "C" void kernel_launch(void* const* d_in, const int* in_sizes, int n_in,
                              void* d_out, int out_size, void* d_ws, size_t ws_size,
                              hipStream_t stream) {
  (void)in_sizes; (void)n_in; (void)out_size; (void)ws_size;
  const float* inp    = (const float*)d_in[0];
  const float* ln_g   = (const float*)d_in[1];
  const float* ln_b   = (const float*)d_in[2];
  const float* wqkv   = (const float*)d_in[3];
  const float* wout   = (const float*)d_in[4];
  const float* gate_w = (const float*)d_in[5];
  const float* gate_b = (const float*)d_in[6];
  const float* b1_kw1 = (const float*)d_in[7];
  const float* b1_kb1 = (const float*)d_in[8];
  const float* b1_kw2 = (const float*)d_in[9];
  const float* b1_kb2 = (const float*)d_in[10];
  const float* b1_vw1 = (const float*)d_in[11];
  const float* b1_vb1 = (const float*)d_in[12];
  const float* b1_vw2 = (const float*)d_in[13];
  const float* b1_vb2 = (const float*)d_in[14];
  const float* b2_kw1 = (const float*)d_in[15];
  const float* b2_kb1 = (const float*)d_in[16];
  const float* b2_kw2 = (const float*)d_in[17];
  const float* b2_kb2 = (const float*)d_in[18];
  const float* b2_vw1 = (const float*)d_in[19];
  const float* b2_vb1 = (const float*)d_in[20];
  const float* b2_vw2 = (const float*)d_in[21];
  const float* b2_vb2 = (const float*)d_in[22];

  char* ws = (char*)d_ws;
  size_t off = 0;
  auto alloc = [&](size_t bytes) -> char* {
    char* p = ws + off;
    off += (bytes + 255) & ~(size_t)255;
    return p;
  };

  _Float16* wqkv_h = (_Float16*)alloc((size_t)3072 * 1024 * 2);
  _Float16* wout_h = (_Float16*)alloc((size_t)1024 * 1024 * 2);
  float*    x_ln   = (float*)alloc((size_t)NTOK * EMB * 4);
  _Float16* x_h    = (_Float16*)alloc((size_t)NTOK * EMB * 2);
  float*    qkv    = (float*)alloc((size_t)NTOK * 3072 * 4);
  float*    qf     = (float*)alloc(BHLD * 4);
  float*    kf     = (float*)alloc(BHLD * 4);
  float*    vf     = (float*)alloc(BHLD * 4);
  _Float16* qh     = (_Float16*)alloc(BHLD * 2);
  _Float16* kh     = (_Float16*)alloc(BHLD * 2);
  _Float16* vth    = (_Float16*)alloc(BHLD * 2);
  float*    glb    = (float*)alloc(BHLD * 4);
  float*    kc1    = (float*)alloc((size_t)BH * 64 * 32 * 4);
  float*    vc1    = (float*)alloc((size_t)BH * 64 * 32 * 4);
  float*    kc2    = (float*)alloc((size_t)BH * 256 * 32 * 4);
  float*    vc2    = (float*)alloc((size_t)BH * 256 * 32 * 4);
  float*    bs1    = (float*)alloc((size_t)BH * 64 * 4);
  float*    bs2    = (float*)alloc((size_t)BH * 256 * 4);
  int*      idx1   = (int*)alloc((size_t)BH * 2 * 4);
  int*      idx2   = (int*)alloc((size_t)BH * 8 * 4);
  float*    out1   = (float*)alloc(BHLD * 4);
  float*    out2   = (float*)alloc(BHLD * 4);
  float*    gates  = (float*)alloc((size_t)NTOK * 48 * 4);
  float*    comb   = (float*)alloc((size_t)NTOK * EMB * 4);
  _Float16* y_h    = (_Float16*)alloc((size_t)NTOK * EMB * 2);

  // weight conversions
  f32_to_f16_kernel<<<(3072 * 1024) / 256, 256, 0, stream>>>(wqkv, wqkv_h, 3072 * 1024);
  f32_to_f16_kernel<<<(1024 * 1024) / 256, 256, 0, stream>>>(wout, wout_h, 1024 * 1024);

  // input layer norm
  ln_kernel<<<NTOK, 256, 0, stream>>>(inp, ln_g, ln_b, x_ln, x_h);

  // qkv projection: [4096,1024] @ [3072,1024]^T
  gemm_nt_f16_kernel<<<dim3(NTOK / 16, 3072 / 256), 128, 0, stream>>>(
      x_h, wqkv_h, qkv, NTOK, 3072, EMB);

  qkv_transform_kernel<<<(NTOK * EMB) / 256, 256, 0, stream>>>(
      qkv, qf, kf, vf, qh, kh, vth);

  // global causal attention
  flash_attn_kernel<<<(BH * (LSEQ / 16)) / 4, 128, 0, stream>>>(qh, kh, vth, glb);

  // compression branches
  hipMemsetAsync(bs1, 0, (size_t)BH * 64 * 4, stream);
  hipMemsetAsync(bs2, 0, (size_t)BH * 256 * 4, stream);

  comp_mlp_kernel<<<BH * 64, 32, 0, stream>>>(kf, b1_kw1, b1_kb1, b1_kw2, b1_kb2, kc1, 32, 64);
  comp_mlp_kernel<<<BH * 64, 32, 0, stream>>>(vf, b1_vw1, b1_vb1, b1_vw2, b1_vb2, vc1, 32, 64);
  comp_mlp_kernel<<<BH * 256, 32, 0, stream>>>(kf, b2_kw1, b2_kb1, b2_kw2, b2_kb2, kc2, 8, 256);
  comp_mlp_kernel<<<BH * 256, 32, 0, stream>>>(vf, b2_vw1, b2_vb1, b2_vw2, b2_vb2, vc2, 8, 256);

  dim3 gca(LSEQ / 256, BH);
  comp_attn_kernel<<<gca, 256, (64 * 32 + 64) * 4, stream>>>(qf, kc1, vc1, out1, bs1, 32, 64);
  comp_attn_kernel<<<gca, 256, (256 * 32 + 256) * 4, stream>>>(qf, kc2, vc2, out2, bs2, 8, 256);

  topk_kernel<<<BH, 32, 0, stream>>>(bs1, idx1, 64, 2);
  topk_kernel<<<BH, 32, 0, stream>>>(bs2, idx2, 256, 8);

  sel_attn_kernel<<<gca, 256, 0, stream>>>(qf, kf, vf, idx1, out1, 32, 2);
  sel_attn_kernel<<<gca, 256, 0, stream>>>(qf, kf, vf, idx2, out2, 8, 8);

  // gates, combine, output LN, output projection
  gate_kernel<<<NTOK, 64, 0, stream>>>(x_ln, gate_w, gate_b, gates);
  combine_kernel<<<(NTOK * EMB) / 256, 256, 0, stream>>>(out1, out2, glb, gates, comb);
  ln_kernel<<<NTOK, 256, 0, stream>>>(comb, ln_g, ln_b, nullptr, y_h);
  gemm_nt_f16_kernel<<<dim3(NTOK / 16, 1024 / 256), 128, 0, stream>>>(
      y_h, wout_h, (float*)d_out, NTOK, EMB, EMB);
}
